// Model_25975962206896
// MI455X (gfx1250) — matrix operand order repaired and checked by reading, not compiled, not run on hardware
//
#include <hip/hip_runtime.h>
#include <hip/hip_bf16.h>

// ---------------------------------------------------------------------------
// Quantum-circuit batch evaluator for MI455X (gfx1250, wave32).
//
// The 6-layer circuit is a fixed 16x16 complex unitary W (independent of the
// batch).  P(q0=0) needs only rows 0..7 of W.  Realified:
//   O[B,16] = X[B,32] @ Wr[32,16],  p0 = sum_n O[row,n]^2
// which is exactly v_wmma_f32_16x16x32_f16 (M=16 rows/wave, K=32, N=16).
// HBM traffic ~6MB (~0.26us @ 23.3TB/s).  Trig uses the hardware
// v_sin_f32/v_cos_f32 path (__sinf/__cosf): args are bounded to [0, pi], so
// libm's Payne-Hanek slow path is pure overhead (it dominated round 2's
// instruction mix).
// ---------------------------------------------------------------------------

typedef __attribute__((ext_vector_type(16))) _Float16 v16h;
typedef __attribute__((ext_vector_type(8)))  float    v8f;

struct cplx { float x, y; };
__device__ __forceinline__ cplx cmul(cplx a, cplx b) {
    return { a.x * b.x - a.y * b.y, a.x * b.y + a.y * b.x };
}
__device__ __forceinline__ cplx cadd(cplx a, cplx b) {
    return { a.x + b.x, a.y + b.y };
}

union AFrag { v16h v; _Float16 h[16]; };
union DFrag { v8f  v; float    f[8];  };

// ds_swizzle group-of-32 mode: imm = {0, xor[14:10], or[9:5], and[4:0]}.
// and_mask=0x1f keeps all 5 lane bits; xor of bits 0..3 stays within each
// 16-lane half (lanes 0-15 / 16-31), matching the WMMA C/D layout halves.
#define SWZ_XOR(x, m) \
    __int_as_float(__builtin_amdgcn_ds_swizzle(__float_as_int(x), (0x1f | ((m) << 10))))

// ---------------------------------------------------------------------------
// Stage 1: build W (16x16 complex), realify rows 0..7 into Wr[32][16], pack
// the per-lane WMMA B fragment into d_ws (32 lanes x 16 halves).
// B layout used here (contiguous K order, same as A): lane n<16: half h = Wr[K=h][N=n];
// lane n>=16: half h = Wr[K=16+h][N=n-16].
// ---------------------------------------------------------------------------
__global__ void qc_build_w(const float* __restrict__ wU,
                           const float* __restrict__ wR,
                           _Float16* __restrict__ bfrag) {
    __shared__ float WrS[32][16];
    const int t = threadIdx.x;          // 0..31
    if (t < 16) {
        // Thread t owns column t (gates left-multiply -> columns independent).
        cplx col[16];
        #pragma unroll
        for (int r = 0; r < 16; ++r) col[r] = { (r == t) ? 1.f : 0.f, 0.f };

        const int pairsP[6] = {1, 1, 1, 1, 0, 0};
        const int pairsQ[6] = {2, 3, 2, 3, 3, 2};
        // Flattened index i*8+j*4+k*2+l: qubit q lives at bit (3-q).
        for (int l = 0; l < 6; ++l) {
            float th = wU[l];
            float ct = __cosf(th), st = __sinf(th);
            int bp = 1 << (3 - pairsP[l]);
            int bq = 1 << (3 - pairsQ[l]);
            cplx is = { 0.f, st };
            #pragma unroll
            for (int r = 0; r < 16; ++r) {
                if (r & (bp | bq)) continue;
                int r01 = r | bq, r10 = r | bp;
                cplx a = col[r01], b = col[r10];
                col[r01] = cadd({ ct * a.x, ct * a.y }, cmul(is, b));
                col[r10] = cadd(cmul(is, a), { ct * b.x, ct * b.y });
            }
            for (int qb = 0; qb < 4; ++qb) {
                const float aA = wR[(l * 4 + qb) * 3 + 0];
                const float bB = wR[(l * 4 + qb) * 3 + 1];
                const float cC = wR[(l * 4 + qb) * 3 + 2];
                float ca = __cosf(0.5f * aA), sa = __sinf(0.5f * aA);
                float cb = __cosf(0.5f * bB), sb = __sinf(0.5f * bB);
                float cc = __cosf(0.5f * cC), sc = __sinf(0.5f * cC);
                cplx rx1[2][2] = { { {ca,0.f}, {0.f,-sa} }, { {0.f,-sa}, {ca,0.f} } };
                cplx rz [2][2] = { { {cb,-sb}, {0.f,0.f} }, { {0.f,0.f}, {cb, sb} } };
                cplx rx2[2][2] = { { {cc,0.f}, {0.f,-sc} }, { {0.f,-sc}, {cc,0.f} } };
                cplx tmp[2][2], g[2][2];
                #pragma unroll
                for (int i = 0; i < 2; ++i)
                    #pragma unroll
                    for (int j = 0; j < 2; ++j)
                        tmp[i][j] = cadd(cmul(rz[i][0], rx2[0][j]),
                                         cmul(rz[i][1], rx2[1][j]));
                #pragma unroll
                for (int i = 0; i < 2; ++i)
                    #pragma unroll
                    for (int j = 0; j < 2; ++j)
                        g[i][j] = cadd(cmul(rx1[i][0], tmp[0][j]),
                                       cmul(rx1[i][1], tmp[1][j]));
                int bit = 1 << (3 - qb);
                #pragma unroll
                for (int r = 0; r < 16; ++r) {
                    if (r & bit) continue;
                    int r1 = r | bit;
                    cplx a0 = col[r], a1 = col[r1];
                    col[r ] = cadd(cmul(g[0][0], a0), cmul(g[0][1], a1));
                    col[r1] = cadd(cmul(g[1][0], a0), cmul(g[1][1], a1));
                }
            }
        }
        // col[m] = W[m][t]: Wr[2t][2m]=Re, Wr[2t+1][2m]=-Im,
        //                   Wr[2t][2m+1]=Im, Wr[2t+1][2m+1]=Re
        #pragma unroll
        for (int m = 0; m < 8; ++m) {
            float re = col[m].x, im = col[m].y;
            WrS[2 * t    ][2 * m    ] =  re;
            WrS[2 * t + 1][2 * m    ] = -im;
            WrS[2 * t    ][2 * m + 1] =  im;
            WrS[2 * t + 1][2 * m + 1] =  re;
        }
    }
    __syncthreads();
    const int N    = t & 15;
    const int kOff = (t < 16) ? 0 : 16;
    #pragma unroll
    for (int h = 0; h < 16; ++h)
        bfrag[t * 16 + h] = (_Float16)WrS[kOff + h][N];
}

// ---------------------------------------------------------------------------
// Stage 2: two 16-row tiles per wave (independent work fills the WMMA->VALU
// hazard slots).  Lanes L and L+16 share batch row M=L; each builds its half
// of the A fragment (contiguous K order, same as B: lane-lo K{0..15} ->
// amps {0..7}; lane-hi K{16..31} -> amps {8..15}).
// Row reduction: exchange-and-fold with 8 raw ds_swizzle ops; afterwards
// lane n holds p0 for tile-row (n>>1); even lanes store one float each.
// ---------------------------------------------------------------------------
#define TILES_PER_WAVE 2

__global__ __launch_bounds__(256) void qc_batch_kernel(
        const float* __restrict__ in,        // [B,2] angles
        const _Float16* __restrict__ bfrag,  // [32][16] B fragment
        const float* __restrict__ sScale,
        const float* __restrict__ sBias,
        float* __restrict__ out, int nTiles) {
    const int lane = threadIdx.x & 31;
    const int wave = blockIdx.x * (blockDim.x >> 5) + (threadIdx.x >> 5);
    const int tile0 = wave * TILES_PER_WAVE;
    if (tile0 >= nTiles) return;         // whole-wave uniform (WMMA needs EXEC=~0)

    const int  M  = lane & 15;
    const bool hi = lane >= 16;
    const int  pa = hi ? 2 : 0;          // amps p*4+q: lo needs p=0,1; hi p=2,3
    const int  pb = hi ? 3 : 1;

    const v16h bm = ((const v16h*)bfrag)[lane];
    const float scale = *sScale;
    const float bias  = *sBias;
    const float kNoise = 0.044721359549995794f;   // sqrt(2/1000)

    DFrag d[TILES_PER_WAVE];
    #pragma unroll
    for (int tt = 0; tt < TILES_PER_WAVE; ++tt) {
        const int b = (tile0 + tt) * 16 + M;
        // Encoding: v = (RZ(f0)RY(f1))|0> ; state = v^{(x)4}
        const float2 f = ((const float2*)in)[b];
        float c0 = __cosf(0.5f * f.x), s0 = __sinf(0.5f * f.x);
        float c1 = __cosf(0.5f * f.y), s1 = __sinf(0.5f * f.y);
        cplx v0 = { c0 * c1, -s0 * c1 };
        cplx v1 = { c0 * s1,  s0 * s1 };
        cplx vv[4];
        vv[0] = cmul(v0, v0); vv[1] = cmul(v0, v1);
        vv[2] = vv[1];        vv[3] = cmul(v1, v1);

        AFrag a;
        #pragma unroll
        for (int q = 0; q < 4; ++q) {
            cplx am = cmul(vv[pa], vv[q]);
            a.h[2 * q    ] = (_Float16)am.x;
            a.h[2 * q + 1] = (_Float16)am.y;
        }
        #pragma unroll
        for (int q = 0; q < 4; ++q) {
            cplx am = cmul(vv[pb], vv[q]);
            a.h[8 + 2 * q    ] = (_Float16)am.x;
            a.h[8 + 2 * q + 1] = (_Float16)am.y;
        }

        v8f cz = {};
        d[tt].v = __builtin_amdgcn_wmma_f32_16x16x32_f16(
            false, a.v, false, bm, (short)0, cz, false, false);
    }

    const bool b3 = (lane & 8) != 0;
    const bool b2 = (lane & 4) != 0;
    const bool b1 = (lane & 2) != 0;

    #pragma unroll
    for (int tt = 0; tt < TILES_PER_WAVE; ++tt) {
        // Squares, then fold: each step halves the live registers while
        // splitting rows across lane-bit groups.  After the last step,
        // lane n holds the 16-lane row-sum for tile-row (n>>1).
        float sq[8];
        #pragma unroll
        for (int i = 0; i < 8; ++i) sq[i] = d[tt].f[i] * d[tt].f[i];

        float a4[4];
        #pragma unroll
        for (int i = 0; i < 4; ++i) {
            float snd = b3 ? sq[i]     : sq[i + 4];
            float kep = b3 ? sq[i + 4] : sq[i];
            a4[i] = kep + SWZ_XOR(snd, 8);
        }
        float a2[2];
        #pragma unroll
        for (int i = 0; i < 2; ++i) {
            float snd = b2 ? a4[i]     : a4[i + 2];
            float kep = b2 ? a4[i + 2] : a4[i];
            a2[i] = kep + SWZ_XOR(snd, 4);
        }
        float snd = b1 ? a2[0] : a2[1];
        float kep = b1 ? a2[1] : a2[0];
        float a1 = kep + SWZ_XOR(snd, 2);
        float p0 = a1 + SWZ_XOR(a1, 1);

        float Z = 2.f * p0 - 1.f;
        float res = scale * (Z + kNoise * (Z * Z - 1.f) * 0.25f) + bias;
        if ((lane & 1) == 0)
            out[(tile0 + tt) * 16 + (lane >> 1)] = res;
    }
}

extern "C" void kernel_launch(void* const* d_in, const int* in_sizes, int n_in,
                              void* d_out, int out_size, void* d_ws, size_t ws_size,
                              hipStream_t stream) {
    const float* inputs = (const float*)d_in[0];   // [B,2]
    const float* wU     = (const float*)d_in[1];   // [6]
    const float* wR     = (const float*)d_in[2];   // [6,4,3]
    const float* oScale = (const float*)d_in[3];   // [1]
    const float* oBias  = (const float*)d_in[4];   // [1]
    float*       out    = (float*)d_out;           // [B] float32
    _Float16*    bfrag  = (_Float16*)d_ws;         // 1 KiB B-fragment

    qc_build_w<<<1, 32, 0, stream>>>(wU, wR, bfrag);

    const int nTiles = out_size / 16;                        // 32768 for B=524288
    const int waves  = (nTiles + TILES_PER_WAVE - 1) / TILES_PER_WAVE;
    const int blocks = (waves + 7) / 8;                      // 8 waves per block
    qc_batch_kernel<<<blocks, 256, 0, stream>>>(inputs, bfrag, oScale, oBias,
                                                out, nTiles);
}
